// PixelDistributionLoss_33861522162284
// MI455X (gfx1250) — compile-verified, matrix operand order checked by reading, not run on hardware
//
#include <hip/hip_runtime.h>
#include <hip/hip_bf16.h>
#include <math.h>

typedef __attribute__((ext_vector_type(16))) _Float16 v16h;
typedef __attribute__((ext_vector_type(8)))  float    v8f;

#define BINS        256
#define NPIX        16384      // 128*128 per channel
#define NCH         24         // 8*3
#define CHUNKS      8
#define CHUNK_PIX   (NPIX / CHUNKS)          // 2048
#define HIST_BLOCKS (2 * NCH * CHUNKS)       // 384
#define INV_SIGMA   100.0f                   // 1/0.01
// exp(-0.5*t^2) = exp2(-0.5*log2(e)*t^2)
#define NEG_HALF_LOG2E (-0.72134752044448170f)
#define INV_NORM    39.894228040143274f      // 1/(sigma*sqrt(2*pi)), sigma=0.01
#define WS_KL_OFF   (2 * NCH * CHUNKS * BINS)  // float offset of per-channel KL region

// ---------------------------------------------------------------------------
// Kernel 1: WMMA soft-histogram partials.
// grid = 384 blocks (t, ch, chunk), block = 512 threads = 16 wave32s.
// Wave w owns bins [16w, 16w+16); lane L owns bin column n = L&15 and
// pixel K-half (L>>4). A = all-ones f16, so D[*, n] = sum_k B[k, n]:
// one v_wmma_f32_16x16x32_f16 reduces 32 pixels x 16 bins per issue.
// ---------------------------------------------------------------------------
__global__ __launch_bounds__(512)
void soft_hist_wmma_kernel(const float* __restrict__ target,
                           const float* __restrict__ pred,
                           float* __restrict__ ws_part) {
    __shared__ __align__(16) float tile[CHUNK_PIX];   // 8 KB

    const int bid   = blockIdx.x;                 // [0, 384)
    const int t     = bid / (NCH * CHUNKS);       // 0 = target, 1 = prediction
    const int rem   = bid % (NCH * CHUNKS);
    const int ch    = rem / CHUNKS;
    const int chunk = rem % CHUNKS;

    const float* __restrict__ src =
        (t == 0 ? target : pred) + ch * NPIX + chunk * CHUNK_PIX;

    // Cooperative stage of the 2048-pixel chunk into LDS.
    for (int i = threadIdx.x; i < CHUNK_PIX; i += 512)
        tile[i] = src[i];
    __syncthreads();

    const int wave = threadIdx.x >> 5;            // [0,16)
    const int lane = threadIdx.x & 31;
    const int bin  = wave * 16 + (lane & 15);
    const float center = (float)bin * (1.0f / 255.0f);

    // A matrix: all ones.
    v16h a;
#pragma unroll
    for (int j = 0; j < 16; ++j) a[j] = (_Float16)1.0f;

    v8f acc = {};

    // 64 iterations of 32 pixels each; lane group selects which 16.
    for (int it = 0; it < CHUNK_PIX / 32; ++it) {
        const int base = it * 32 + ((lane >> 4) << 4);
        const float4* __restrict__ p4 =
            reinterpret_cast<const float4*>(&tile[base]);
        float px[16];
#pragma unroll
        for (int q = 0; q < 4; ++q) {
            float4 v = p4[q];                     // ds_load_b128 (broadcast)
            px[4 * q + 0] = v.x;
            px[4 * q + 1] = v.y;
            px[4 * q + 2] = v.z;
            px[4 * q + 3] = v.w;
        }
        v16h b;
#pragma unroll
        for (int j = 0; j < 16; ++j) {
            float d = (px[j] - center) * INV_SIGMA;
            float w = __builtin_amdgcn_exp2f(NEG_HALF_LOG2E * d * d);
            b[j] = (_Float16)w;
        }
        acc = __builtin_amdgcn_wmma_f32_16x16x32_f16(
            false, a, false, b, (short)0, acc, false, false);
    }

    // Every row of D equals the column sum; lanes 0..15 hold columns 0..15.
    if (lane < 16)
        ws_part[bid * BINS + bin] = acc[0];
}

// ---------------------------------------------------------------------------
// Kernel 2: per-channel normalize + KL. grid = 24 blocks, 256 threads (= bins).
// Fixed-order chunk sums and fixed LDS reduction trees => deterministic.
// ---------------------------------------------------------------------------
__global__ __launch_bounds__(256)
void kl_kernel(const float* __restrict__ ws_part,
               float* __restrict__ ws_kl) {
    __shared__ float red[256];
    const int c = blockIdx.x;        // channel [0,24)
    const int b = threadIdx.x;       // bin

    float traw = 0.0f, praw = 0.0f;
#pragma unroll
    for (int k = 0; k < CHUNKS; ++k)
        traw += ws_part[((0 * NCH + c) * CHUNKS + k) * BINS + b];
#pragma unroll
    for (int k = 0; k < CHUNKS; ++k)
        praw += ws_part[((1 * NCH + c) * CHUNKS + k) * BINS + b];

    const float tv = traw * INV_NORM;
    const float pv = praw * INV_NORM;

    // --- reduce sum of tv ---
    red[b] = tv; __syncthreads();
    for (int s = 128; s > 0; s >>= 1) {
        if (b < s) red[b] += red[b + s];
        __syncthreads();
    }
    const float sumT = red[0]; __syncthreads();

    // --- reduce sum of pv ---
    red[b] = pv; __syncthreads();
    for (int s = 128; s > 0; s >>= 1) {
        if (b < s) red[b] += red[b + s];
        __syncthreads();
    }
    const float sumP = red[0]; __syncthreads();

    const float p = tv / (sumT + 1e-10f) + 1e-10f;  // target hist
    const float q = pv / (sumP + 1e-10f) + 1e-10f;  // prediction hist
    const float term = p * __logf(p / q);

    // --- reduce sum of KL terms ---
    red[b] = term; __syncthreads();
    for (int s = 128; s > 0; s >>= 1) {
        if (b < s) red[b] += red[b + s];
        __syncthreads();
    }
    if (b == 0) ws_kl[c] = red[0];
}

// ---------------------------------------------------------------------------
// Kernel 3: mean over 24 channels, fixed order, single thread.
// ---------------------------------------------------------------------------
__global__ void mean_kernel(const float* __restrict__ ws_kl,
                            float* __restrict__ out) {
    if (threadIdx.x == 0 && blockIdx.x == 0) {
        float s = 0.0f;
        for (int c = 0; c < NCH; ++c) s += ws_kl[c];
        out[0] = s / (float)NCH;
    }
}

extern "C" void kernel_launch(void* const* d_in, const int* in_sizes, int n_in,
                              void* d_out, int out_size, void* d_ws, size_t ws_size,
                              hipStream_t stream) {
    (void)in_sizes; (void)n_in; (void)out_size; (void)ws_size;
    const float* target = (const float*)d_in[0];
    const float* pred   = (const float*)d_in[1];
    float* ws_part = (float*)d_ws;                  // 2*24*8*256 floats
    float* ws_kl   = ws_part + WS_KL_OFF;           // 24 floats
    float* out     = (float*)d_out;                 // 1 float

    soft_hist_wmma_kernel<<<HIST_BLOCKS, 512, 0, stream>>>(target, pred, ws_part);
    kl_kernel<<<NCH, 256, 0, stream>>>(ws_part, ws_kl);
    mean_kernel<<<1, 32, 0, stream>>>(ws_kl, out);
}
